// CausalAttention_56444460204439
// MI455X (gfx1250) — compile-verified, matrix operand order checked by reading, not run on hardware
//
#include <hip/hip_runtime.h>

// ---------------------------------------------------------------------------
// CausalAttention for MI455X (gfx1250, wave32, WMMA bf16 + TDM staging)
// B=4, T=2048, D_IN=D_OUT=1024
// ---------------------------------------------------------------------------

#define B_   4
#define T_   2048
#define D_   1024
#define ROWS (B_ * T_)          // 8192 rows of X

typedef __attribute__((ext_vector_type(16))) __bf16 v16bf;
typedef __attribute__((ext_vector_type(8)))  float  v8f;
typedef __attribute__((ext_vector_type(8)))  unsigned short v8us;
typedef __attribute__((ext_vector_type(4)))  unsigned short v4us;
typedef __attribute__((ext_vector_type(4)))  unsigned int   v4u;
typedef __attribute__((ext_vector_type(4)))  int            v4i;
typedef __attribute__((ext_vector_type(8)))  int            v8i;

// ---------------------------------------------------------------------------
// Tensor Data Mover: 2D bf16 tile load Global -> LDS.
// D# built per CDNA5 ISA 8.3/8.4: group0 = {count/flags, lds_addr,
// global_addr lo, global_addr hi | type=2}; group1 packs data_size=2B,
// tensor_dim0/1, tile_dim0/1, tensor_dim0_stride. Groups 2/3 zero (2D tile).
// Issued by one wave; drained with s_wait_tensorcnt.
// ---------------------------------------------------------------------------
#if __has_builtin(__builtin_amdgcn_tensor_load_to_lds) && \
    __has_builtin(__builtin_amdgcn_s_wait_tensorcnt)
#define HAVE_TDM 1
__device__ __forceinline__ void tdm_load_2d_bf16(
    unsigned lds_byte_off, const void* gaddr,
    unsigned tensor_w, unsigned tensor_h,     // tensor dim0 / dim1 (elements)
    unsigned tile_w, unsigned tile_h,         // tile dim0 / dim1 (elements)
    unsigned row_stride)                      // tensor dim0 stride (elements)
{
    unsigned long long ga = (unsigned long long)gaddr;
    v4u g0;
    g0.x = 1u;                                            // count=1, user desc
    g0.y = lds_byte_off;                                  // lds_addr (bytes)
    g0.z = (unsigned)(ga & 0xFFFFFFFFull);                // global_addr[31:0]
    g0.w = (unsigned)((ga >> 32) & 0x1FFFFFFull)          // global_addr[56:32]
         | (2u << 30);                                    // type=2 ("image")
    v8i g1;
    g1[0] = (int)(1u << 16);                              // data_size=1 -> 2B
    g1[1] = (int)(tensor_w << 16);                        // tensor_dim0[15:0]
    g1[2] = (int)((tensor_w >> 16) | (tensor_h << 16));   // dim0 hi | dim1 lo
    g1[3] = (int)((tensor_h >> 16) | (tile_w << 16));     // dim1 hi | tile_dim0
    g1[4] = (int)(tile_h & 0xFFFFu);                      // tile_dim1 (dim2=0)
    g1[5] = (int)row_stride;                              // dim0_stride[31:0]
    g1[6] = 0;
    g1[7] = 0;
    v4i gz = {0, 0, 0, 0};
#if __clang_major__ >= 23
    v8i gz8 = {0, 0, 0, 0, 0, 0, 0, 0};
    __builtin_amdgcn_tensor_load_to_lds(g0, g1, gz, gz, gz8, 0);
#else
    __builtin_amdgcn_tensor_load_to_lds(g0, g1, gz, gz, 0);
#endif
}
#else
#define HAVE_TDM 0
#endif

// round-to-nearest-even fp32 -> bf16 (raw ushort)
__device__ __forceinline__ unsigned short f2bf(float f) {
    union { float f; unsigned u; } x; x.f = f;
    unsigned u = x.u + 0x7FFFu + ((x.u >> 16) & 1u);
    return (unsigned short)(u >> 16);
}

// Load one 16x32 bf16 WMMA fragment (A-layout; B uses the same lane->N map).
// Lane L: row = L%16; K-chunks at 8*(L>=16) and 16+8*(L>=16), each 8
// contiguous bf16 (16B) -> matches CDNA5 ISA 16-bit A/B layout.
__device__ __forceinline__ v16bf load_frag(const unsigned short* p, int stride, int lane) {
    const unsigned short* q = p + (lane & 15) * stride + ((lane >> 4) & 1) * 8;
    struct Pair { uint4 lo, hi; } pr;
    pr.lo = *reinterpret_cast<const uint4*>(q);
    pr.hi = *reinterpret_cast<const uint4*>(q + 16);
    return __builtin_bit_cast(v16bf, pr);
}

// ---------------------------------------------------------------------------
// Stage 1: fp32 -> bf16 conversion (vectorized x4)
// ---------------------------------------------------------------------------
__global__ void __launch_bounds__(256) cvt_f32_bf16(const float* __restrict__ src,
                                                    unsigned short* __restrict__ dst,
                                                    int n4) {
    int i = blockIdx.x * blockDim.x + threadIdx.x;
    if (i < n4) {
        float4 f = reinterpret_cast<const float4*>(src)[i];
        v4us o;
        o.x = f2bf(f.x); o.y = f2bf(f.y); o.z = f2bf(f.z); o.w = f2bf(f.w);
        reinterpret_cast<v4us*>(dst)[i] = o;
    }
}

// ---------------------------------------------------------------------------
// Stage 2: QKV projection  Out[r, n] = sum_d X[r, d] * W[n, d]
// 256 threads (8 waves). Block tile 128x128, wave tile 64x32 (4x2 WMMA tiles).
// Dynamic LDS: Xs @ +0 (128x32 bf16), Ws @ +8192 (128x32 bf16).
// grid = (ROWS/128, D/128, 3) with z selecting Wq/Wk/Wv.
// ---------------------------------------------------------------------------
__global__ void __launch_bounds__(256) gemm_qkv(
    const unsigned short* __restrict__ Xh,
    const unsigned short* __restrict__ Wqh,
    const unsigned short* __restrict__ Wkh,
    const unsigned short* __restrict__ Wvh,
    unsigned short* __restrict__ Qh,
    unsigned short* __restrict__ Kh,
    unsigned short* __restrict__ Vh) {

    extern __shared__ __align__(16) char gsm[];
    unsigned short* Xs = reinterpret_cast<unsigned short*>(gsm);          // 128*32
    unsigned short* Ws = Xs + 128 * 32;                                   // 128*32
    const unsigned ldsbase = __builtin_amdgcn_groupstaticsize();

    const unsigned short* W  = (blockIdx.z == 0) ? Wqh : (blockIdx.z == 1) ? Wkh : Wvh;
    unsigned short*       Ot = (blockIdx.z == 0) ? Qh  : (blockIdx.z == 1) ? Kh  : Vh;

    const int m0   = blockIdx.x * 128;
    const int n0   = blockIdx.y * 128;
    const int tid  = threadIdx.x;
    const int lane = tid & 31;
    const int w    = tid >> 5;
    const int wm   = w >> 2;   // 0..1 : 64-row slab
    const int wn   = w & 3;    // 0..3 : 32-col slab

    v8f acc[4][2];
#pragma unroll
    for (int i = 0; i < 4; ++i)
#pragma unroll
        for (int j = 0; j < 2; ++j) acc[i][j] = (v8f)(0.0f);

    for (int k0 = 0; k0 < D_; k0 += 32) {
#if HAVE_TDM
        // TDM: DMA 128x32 tiles of X and W into LDS (wave 0 issues; EXEC-free)
        if (w == 0) {
            tdm_load_2d_bf16(ldsbase + 0,
                             Xh + (size_t)m0 * D_ + k0,
                             D_, (unsigned)(ROWS - m0), 32, 128, D_);
            tdm_load_2d_bf16(ldsbase + 128 * 32 * 2,
                             W + (size_t)n0 * D_ + k0,
                             D_, (unsigned)(D_ - n0), 32, 128, D_);
            __builtin_amdgcn_s_wait_tensorcnt(0);
        }
        if (k0 + 32 < D_) {   // prefetch next K-slab (global_prefetch_b8)
            __builtin_prefetch(Xh + (size_t)(m0 + (tid >> 1)) * D_ + k0 + 32, 0, 0);
            __builtin_prefetch(W  + (size_t)(n0 + (tid >> 1)) * D_ + k0 + 32, 0, 0);
        }
#else
        {
            const int r = tid >> 1;
            const int c = (tid & 1) * 16;
            const uint4* xs = reinterpret_cast<const uint4*>(Xh + (size_t)(m0 + r) * D_ + k0 + c);
            uint4* xd = reinterpret_cast<uint4*>(Xs + r * 32 + c);
            xd[0] = xs[0]; xd[1] = xs[1];
            const uint4* wsrc = reinterpret_cast<const uint4*>(W + (size_t)(n0 + r) * D_ + k0 + c);
            uint4* wd = reinterpret_cast<uint4*>(Ws + r * 32 + c);
            wd[0] = wsrc[0]; wd[1] = wsrc[1];
            if (k0 + 32 < D_) {
                __builtin_prefetch(Xh + (size_t)(m0 + r) * D_ + k0 + 32 + c, 0, 0);
                __builtin_prefetch(W  + (size_t)(n0 + r) * D_ + k0 + 32 + c, 0, 0);
            }
        }
#endif
        __syncthreads();

        v16bf a[4], b[2];
#pragma unroll
        for (int i = 0; i < 4; ++i) a[i] = load_frag(Xs + (wm * 64 + i * 16) * 32, 32, lane);
#pragma unroll
        for (int j = 0; j < 2; ++j) b[j] = load_frag(Ws + (wn * 32 + j * 16) * 32, 32, lane);

#pragma unroll
        for (int i = 0; i < 4; ++i)
#pragma unroll
            for (int j = 0; j < 2; ++j)
                acc[i][j] = __builtin_amdgcn_wmma_f32_16x16x32_bf16(
                    false, a[i], false, b[j], (short)0, acc[i][j], false, false);
        __syncthreads();
    }

    // C/D layout: VGPR r -> M = r + 8*(lane>=16), N = lane%16
    const int cn = lane & 15;
    const int mh = (lane >> 4) & 1;
#pragma unroll
    for (int i = 0; i < 4; ++i)
#pragma unroll
        for (int j = 0; j < 2; ++j) {
            const int col = n0 + wn * 32 + j * 16 + cn;
#pragma unroll
            for (int r = 0; r < 8; ++r) {
                const int row = m0 + wm * 64 + i * 16 + r + 8 * mh;
                Ot[(size_t)row * D_ + col] = f2bf(acc[i][j][r]);
            }
        }
}

// ---------------------------------------------------------------------------
// Stage 3: fused causal flash attention.
// Block = 256 threads (8 waves) = one 64-query tile of one batch.
// Dynamic LDS layout (byte offsets from groupstaticsize base):
//   Qs @ 0       : 64x1024 bf16 (131072)  - TDM
//   Ks @ 131072  : 32x1024 bf16 ( 65536)  - TDM per key tile
//   Vt @ 196608  : Vt[d][k] 1024x32 bf16  - manual transposed store
//   Ss @ 262144  : 64x32 f32
//   Ps @ 270336  : 64x32 bf16
//   alpha @ 274432, lrow @ 274688 : 64 f32 each        => 274,944 B total
// ---------------------------------------------------------------------------
#define QS_OFF 0u
#define KS_OFF 131072u
#define VT_OFF 196608u

__global__ void __launch_bounds__(256) attn_fused(
    const unsigned short* __restrict__ Qh,
    const unsigned short* __restrict__ Kh,
    const unsigned short* __restrict__ Vh,
    float* __restrict__ Out) {

    extern __shared__ __align__(16) char smem[];
    unsigned short* Qs = reinterpret_cast<unsigned short*>(smem);
    unsigned short* Ks = Qs + 64 * 1024;
    unsigned short* Vt = Ks + 32 * 1024;                              // Vt[d][k]
    float*          Ss = reinterpret_cast<float*>(Vt + 32 * 1024);
    unsigned short* Ps = reinterpret_cast<unsigned short*>(Ss + 64 * 32);
    float*       alpha = reinterpret_cast<float*>(Ps + 64 * 32);
    float*        lrow = alpha + 64;
    const unsigned ldsbase = __builtin_amdgcn_groupstaticsize();

    const int b  = blockIdx.y;
    const int q0 = blockIdx.x * 64;
    const unsigned short* Qb = Qh + (size_t)b * T_ * D_;
    const unsigned short* Kb = Kh + (size_t)b * T_ * D_;
    const unsigned short* Vb = Vh + (size_t)b * T_ * D_;
    float*                Ob = Out + (size_t)b * T_ * D_;

    const int tid  = threadIdx.x;
    const int lane = tid & 31;
    const int w    = tid >> 5;
    const int cn   = lane & 15;
    const int mh   = (lane >> 4) & 1;

    // ---- load Q tile (64 x 1024) into LDS ----
#if HAVE_TDM
    if (w == 0) {
        tdm_load_2d_bf16(ldsbase + QS_OFF, Qb + (size_t)q0 * D_,
                         D_, (unsigned)(T_ - q0), D_, 64, D_);
        __builtin_amdgcn_s_wait_tensorcnt(0);
    }
#else
    for (int idx = tid; idx < 64 * D_ / 8; idx += 256) {
        const int r = idx >> 7;
        const int c = (idx & 127) * 8;
        *reinterpret_cast<uint4*>(Qs + r * D_ + c) =
            *reinterpret_cast<const uint4*>(Qb + (size_t)(q0 + r) * D_ + c);
    }
#endif

    v8f o[4][8];
#pragma unroll
    for (int i = 0; i < 4; ++i)
#pragma unroll
        for (int j = 0; j < 8; ++j) o[i][j] = (v8f)(0.0f);

    float m_i = -__builtin_huge_valf();
    float l_i = 0.0f;

    const int nkt = (q0 >> 5) + 2;   // causal: key tiles covering keys <= q0+63
    __syncthreads();

    for (int kt = 0; kt < nkt; ++kt) {
        const int kk0 = kt * 32;

        // ---- stage K tile (row-major) ----
#if HAVE_TDM
        if (w == 0)
            tdm_load_2d_bf16(ldsbase + KS_OFF, Kb + (size_t)kk0 * D_,
                             D_, (unsigned)(T_ - kk0), D_, 32, D_);
#else
        for (int idx = tid; idx < 32 * D_ / 8; idx += 256) {
            const int r = idx >> 7;
            const int c = (idx & 127) * 8;
            *reinterpret_cast<uint4*>(Ks + r * D_ + c) =
                *reinterpret_cast<const uint4*>(Kb + (size_t)(kk0 + r) * D_ + c);
        }
#endif
        // ---- stage V tile transposed: Vt[d][k] ----
        for (int idx = tid; idx < 32 * D_ / 8; idx += 256) {
            const int r = idx >> 7;
            const int c = (idx & 127) * 8;
            uint4 raw = *reinterpret_cast<const uint4*>(Vb + (size_t)(kk0 + r) * D_ + c);
            v8us vv = __builtin_bit_cast(v8us, raw);
#pragma unroll
            for (int e = 0; e < 8; ++e) Vt[(c + e) * 32 + r] = vv[e];
        }
        if (kt + 1 < nkt) {   // prefetch next tiles into L2/L0
            __builtin_prefetch(Kb + (size_t)(kk0 + 32) * D_ + (size_t)tid * 128, 0, 0);
            __builtin_prefetch(Vb + (size_t)(kk0 + 32) * D_ + (size_t)tid * 128, 0, 0);
        }
#if HAVE_TDM
        if (w == 0) __builtin_amdgcn_s_wait_tensorcnt(0);
#endif
        __syncthreads();

        // ---- S = Q * K^T : 8 waves -> the 4x2 16x16 tiles (Br=64, Bc=32) ----
        {
            const int tm = w & 3;
            const int tn = w >> 2;
            v8f s = (v8f)(0.0f);
            for (int ks = 0; ks < D_; ks += 32) {
                v16bf aq = load_frag(Qs + (tm * 16) * D_ + ks, D_, lane);
                v16bf bk = load_frag(Ks + (tn * 16) * D_ + ks, D_, lane);
                s = __builtin_amdgcn_wmma_f32_16x16x32_bf16(
                        false, aq, false, bk, (short)0, s, false, false);
            }
#pragma unroll
            for (int r = 0; r < 8; ++r)
                Ss[(tm * 16 + r + 8 * mh) * 32 + tn * 16 + cn] = s[r];
        }
        __syncthreads();

        // ---- online softmax over this 32-key slab (one thread per row) ----
        if (tid < 64) {
            const int qrow = q0 + tid;
            float sv[32];
            float mx = -__builtin_huge_valf();
#pragma unroll
            for (int c = 0; c < 32; ++c) {
                float x = Ss[tid * 32 + c] * 0.03125f;     // 1/sqrt(1024)
                if (kk0 + c > qrow) x = -__builtin_huge_valf();
                sv[c] = x;
                mx = fmaxf(mx, x);
            }
            const float mnew = fmaxf(m_i, mx);
            const float al   = __expf(m_i - mnew);
            float sum = 0.0f;
#pragma unroll
            for (int c = 0; c < 32; ++c) {
                const float e = __expf(sv[c] - mnew);
                sum += e;
                Ps[tid * 32 + c] = f2bf(e);
            }
            l_i = l_i * al + sum;
            m_i = mnew;
            alpha[tid] = al;
            if (kt == nkt - 1) lrow[tid] = l_i;
        }
        __syncthreads();

        // ---- O = O*alpha + P @ V_part (inner dim 32 -> single WMMA step) ----
        {
            v16bf pa[4];
#pragma unroll
            for (int i = 0; i < 4; ++i) pa[i] = load_frag(Ps + (i * 16) * 32, 32, lane);
#pragma unroll
            for (int j = 0; j < 8; ++j) {
                v16bf bv = load_frag(Vt + (w * 128 + j * 16) * 32, 32, lane);
#pragma unroll
                for (int i = 0; i < 4; ++i) {
                    v8f a = o[i][j];
#pragma unroll
                    for (int r = 0; r < 8; ++r)
                        a[r] *= alpha[i * 16 + r + 8 * mh];
                    o[i][j] = __builtin_amdgcn_wmma_f32_16x16x32_bf16(
                        false, pa[i], false, bv, (short)0, a, false, false);
                }
            }
        }
        __syncthreads();   // protect Ks/Vt/Ss/Ps for next tile
    }

    // ---- normalize by l and store fp32 ----
#pragma unroll
    for (int i = 0; i < 4; ++i)
#pragma unroll
        for (int r = 0; r < 8; ++r) {
            const int row = i * 16 + r + 8 * mh;
            const float inv = 1.0f / lrow[row];
#pragma unroll
            for (int j = 0; j < 8; ++j)
                Ob[(size_t)(q0 + row) * D_ + w * 128 + j * 16 + cn] = o[i][j][r] * inv;
        }
}

// ---------------------------------------------------------------------------
// Host launcher
// ---------------------------------------------------------------------------
extern "C" void kernel_launch(void* const* d_in, const int* in_sizes, int n_in,
                              void* d_out, int out_size, void* d_ws, size_t ws_size,
                              hipStream_t stream) {
    const float* X  = (const float*)d_in[0];
    const float* Wq = (const float*)d_in[1];
    const float* Wk = (const float*)d_in[2];
    const float* Wv = (const float*)d_in[3];
    float* out = (float*)d_out;

    // workspace layout (bf16 buffers), ~70 MB total
    unsigned short* Xh  = (unsigned short*)d_ws;
    unsigned short* Wqh = Xh  + (size_t)ROWS * D_;
    unsigned short* Wkh = Wqh + (size_t)D_ * D_;
    unsigned short* Wvh = Wkh + (size_t)D_ * D_;
    unsigned short* Qh  = Wvh + (size_t)D_ * D_;
    unsigned short* Kh  = Qh  + (size_t)ROWS * D_;
    unsigned short* Vh  = Kh  + (size_t)ROWS * D_;

    // Stage 1: conversions
    {
        int n4 = ROWS * D_ / 4;
        cvt_f32_bf16<<<(n4 + 255) / 256, 256, 0, stream>>>(X, Xh, n4);
        int w4 = D_ * D_ / 4;
        cvt_f32_bf16<<<(w4 + 255) / 256, 256, 0, stream>>>(Wq, Wqh, w4);
        cvt_f32_bf16<<<(w4 + 255) / 256, 256, 0, stream>>>(Wk, Wkh, w4);
        cvt_f32_bf16<<<(w4 + 255) / 256, 256, 0, stream>>>(Wv, Wvh, w4);
    }

    // Stage 2: QKV projections (z selects which W / output)
    gemm_qkv<<<dim3(ROWS / 128, D_ / 128, 3), 256, 2 * 128 * 32 * 2, stream>>>(
        Xh, Wqh, Wkh, Wvh, Qh, Kh, Vh);

    // Stage 3: fused causal attention
    size_t lds_bytes =
        (size_t)64 * D_ * 2 +   // Qs
        (size_t)32 * D_ * 2 +   // Ks
        (size_t)32 * D_ * 2 +   // Vt
        (size_t)64 * 32 * 4 +   // Ss
        (size_t)64 * 32 * 2 +   // Ps
        64 * 4 + 64 * 4;        // alpha, lrow   => 274,944 B (< 320 KB/WG)
    attn_fused<<<dim3(T_ / 64, B_), 256, lds_bytes, stream>>>(Qh, Kh, Vh, out);
}
